// PromptedWindowAttention_12850542149942
// MI455X (gfx1250) — matrix operand order repaired and checked by reading, not compiled, not run on hardware
//
#include <hip/hip_runtime.h>
#include <hip/hip_bf16.h>
#include <stdint.h>

#define DIM     384
#define HEADS   12
#define HD      32
#define NTOK    54
#define NPROMPT 5
#define WSIDE   49
#define BTOT    2048
#define MROWS   (BTOT * NTOK)   // 110592
#define NWMASK  64

typedef __attribute__((ext_vector_type(16))) __bf16 v16bf;
typedef __attribute__((ext_vector_type(8)))  float  v8f;

union Frag16 {
    uint4 u4[2];
    v16bf v;
};

// round-to-nearest-even float -> bf16 bits
__device__ __forceinline__ unsigned short f2bf(float f) {
    unsigned int u = __float_as_uint(f);
    u += 0x7FFFu + ((u >> 16) & 1u);
    return (unsigned short)(u >> 16);
}

// assemble a 16-element bf16 fragment from two 16-byte chunks
__device__ __forceinline__ v16bf load_frag(const unsigned short* p0,
                                           const unsigned short* p1) {
    Frag16 f;
    f.u4[0] = *(const uint4*)p0;
    f.u4[1] = *(const uint4*)p1;
    return f.v;
}

// ---------------------------------------------------------------------------
// one-shot fp32 -> bf16 conversion, 8 elements / thread
__global__ __launch_bounds__(256)
void cvt_f32_bf16_kernel(const float* __restrict__ src,
                         unsigned short* __restrict__ dst, int n8)
{
    int i = blockIdx.x * 256 + threadIdx.x;
    if (i >= n8) return;
    float4 f0 = ((const float4*)src)[i * 2];
    float4 f1 = ((const float4*)src)[i * 2 + 1];
    __align__(16) unsigned short t[8];
    t[0] = f2bf(f0.x); t[1] = f2bf(f0.y); t[2] = f2bf(f0.z); t[3] = f2bf(f0.w);
    t[4] = f2bf(f1.x); t[5] = f2bf(f1.y); t[6] = f2bf(f1.z); t[7] = f2bf(f1.w);
    ((uint4*)dst)[i] = *(const uint4*)t;
}

// ---------------------------------------------------------------------------
// fused rel-pos-bias + window-mask + key-padding table: bm[w][h][64][64]
__global__ __launch_bounds__(256)
void build_bias_mask_kernel(const float* __restrict__ mask,
                            const float* __restrict__ rpb,
                            const int* __restrict__ rel_index,
                            float* __restrict__ bm)
{
    const int wh = blockIdx.x;            // 0..767 = w*HEADS + h
    const int w = wh / HEADS, h = wh % HEADS;
    for (int e = threadIdx.x; e < 64 * 64; e += 256) {
        const int i = e >> 6, jj = e & 63;
        float v = 0.0f;
        if (jj >= NTOK) {
            v = -1e30f;                   // padded key columns
        } else if (i >= NPROMPT && i < NTOK && jj >= NPROMPT) {
            const int ii = i - NPROMPT, jw = jj - NPROMPT;
            v = rpb[rel_index[ii * WSIDE + jw] * HEADS + h]
              + mask[(size_t)w * WSIDE * WSIDE + ii * WSIDE + jw];
        }
        bm[(size_t)wh * 4096 + e] = v;
    }
}

// ---------------------------------------------------------------------------
// C[M,N] = A[M,K] * W[N,K]^T + bias[N] ; A,W bf16 ; C bf16 or fp32
// Workgroup tile 64x384, 8 waves as 2(m)x4(n), each wave 32x96 = 2x6 WMMA tiles.
template<bool OUT_BF16>
__global__ __launch_bounds__(256)
void gemm_bias_kernel(const unsigned short* __restrict__ A,
                      const unsigned short* __restrict__ W,   // [N][K] bf16
                      const float* __restrict__ bias,
                      void* __restrict__ Cv,
                      int M, int N, int K)
{
    __shared__ unsigned short As[64][40];     // [m][k], stride 80B
    __shared__ unsigned short Bs[384][40];    // [n][k], stride 80B

    const int tid  = threadIdx.x;
    const int lane = tid & 31;
    const int wv   = tid >> 5;
    const int wm   = wv >> 2;            // 0..1
    const int wn   = wv & 3;             // 0..3
    const int m0   = blockIdx.x * 64;
    const int n0   = blockIdx.y * 384;

    const int lm    = lane & 15;
    const int hi16  = lane >> 4;
    const int kselA = hi16 * 8;
    const int kselB = hi16 * 16;

    v8f acc[2][6];
    for (int i = 0; i < 2; i++)
        for (int j = 0; j < 6; j++)
            for (int r = 0; r < 8; r++) acc[i][j][r] = 0.0f;

    const int ar = tid >> 2;             // A row 0..63
    const int ac = (tid & 3) * 8;

    for (int k0 = 0; k0 < K; k0 += 32) {
        // stage A tile: 1 x uint4 per thread
        *(uint4*)&As[ar][ac] = *(const uint4*)(A + (size_t)(m0 + ar) * K + k0 + ac);
        // stage B tile: 6 x uint4 per thread (384 rows x 32 k)
        for (int l = 0; l < 6; l++) {
            const int g   = l * 256 + tid;
            const int row = g >> 2;
            const int off = (g & 3) * 8;
            *(uint4*)&Bs[row][off] =
                *(const uint4*)(W + (size_t)(n0 + row) * K + k0 + off);
        }
        __syncthreads();

        v16bf a[2];
        for (int i = 0; i < 2; i++) {
            const unsigned short* pa = &As[wm * 32 + i * 16 + lm][0];
            a[i] = load_frag(pa + kselA, pa + kselA + 16);
        }
        for (int j = 0; j < 6; j++) {
            const unsigned short* pb = &Bs[wn * 96 + j * 16 + lm][0];
            v16bf bfr = load_frag(pb + kselA, pb + kselA + 16);
            for (int i = 0; i < 2; i++)
                acc[i][j] = __builtin_amdgcn_wmma_f32_16x16x32_bf16(
                    false, a[i], false, bfr, (short)0, acc[i][j], false, false);
        }
        __syncthreads();
    }

    // epilogue: bias + store (C layout: VGPR r -> M = r + hi16*8, N = lane&15)
    for (int i = 0; i < 2; i++)
        for (int j = 0; j < 6; j++) {
            const int nglob = n0 + wn * 96 + j * 16 + lm;
            const float bv  = bias[nglob];
            const int mbase = m0 + wm * 32 + i * 16 + hi16 * 8;
            for (int r = 0; r < 8; r++) {
                float val = acc[i][j][r] + bv;
                size_t idx = (size_t)(mbase + r) * N + nglob;
                if (OUT_BF16) ((unsigned short*)Cv)[idx] = f2bf(val);
                else          ((float*)Cv)[idx] = val;
            }
        }
}

// ---------------------------------------------------------------------------
// One workgroup per (b, head): S = Q K^T -> *scale + bm -> softmax -> O = P V
__global__ __launch_bounds__(256)
void attn_kernel(const unsigned short* __restrict__ qkv,  // [B_,54,3*384] bf16
                 const float* __restrict__ bm,            // [64,12,64,64]
                 unsigned short* __restrict__ attnout)    // [B_,54,384] bf16
{
    __shared__ unsigned short Qs[64][40];   // [n][d]
    __shared__ unsigned short Ks[64][40];   // [m][d]
    __shared__ unsigned short Vt[32][72];   // [d][m] (transposed)
    __shared__ float          Ss[64][65];   // scores
    __shared__ unsigned short Ps[64][72];   // softmax probs bf16

    const int h    = blockIdx.x;
    const int b    = blockIdx.y;
    const int tid  = threadIdx.x;
    const int lane = tid & 31;
    const int wv   = tid >> 5;
    const int lm   = lane & 15;
    const int hi16 = lane >> 4;

    // ---- load Q/K/V head slice, pad rows 54..63 with zeros ----
    {
        const int r  = tid >> 2;          // 0..63
        const int d0 = (tid & 3) * 8;
        if (r < NTOK) {
            const size_t base = ((size_t)b * NTOK + r) * (3 * DIM) + h * HD + d0;
            *(uint4*)&Qs[r][d0] = *(const uint4*)(qkv + base);
            *(uint4*)&Ks[r][d0] = *(const uint4*)(qkv + base + DIM);
            __align__(16) unsigned short t[8];
            *(uint4*)t = *(const uint4*)(qkv + base + 2 * DIM);
            for (int i = 0; i < 8; i++) Vt[d0 + i][r] = t[i];
        } else {
            uint4 z = {0u, 0u, 0u, 0u};
            *(uint4*)&Qs[r][d0] = z;
            *(uint4*)&Ks[r][d0] = z;
            for (int i = 0; i < 8; i++) Vt[d0 + i][r] = 0;
        }
    }
    __syncthreads();

    // ---- S = Q * K^T : 16 tiles of 16x16, 2 per wave ----
    for (int t = wv * 2; t < wv * 2 + 2; t++) {
        const int tm = t >> 2, tn = t & 3;
        const unsigned short* pq = &Qs[tm * 16 + lm][0];
        v16bf a  = load_frag(pq + hi16 * 8, pq + hi16 * 8 + 16);
        const unsigned short* pk = &Ks[tn * 16 + lm][0];
        v16bf bb = load_frag(pk + hi16 * 8, pk + hi16 * 8 + 16);
        v8f c; for (int i = 0; i < 8; i++) c[i] = 0.0f;
        c = __builtin_amdgcn_wmma_f32_16x16x32_bf16(
                false, a, false, bb, (short)0, c, false, false);
        for (int i = 0; i < 8; i++)
            Ss[tm * 16 + hi16 * 8 + i][tn * 16 + lm] = c[i];
    }
    __syncthreads();

    // ---- branch-free scale + fused bias/mask + softmax ----
    {
        const int row = tid >> 2;            // 0..63
        const int c0  = (tid & 3) * 16;      // quad of lanes covers one row
        const float scale = 0.17677669529663687f;  // 32^-0.5
        const float* bmrow =
            bm + ((((size_t)(b & (NWMASK - 1)) * HEADS + h) * 64 + row) * 64);
        float vals[16];
        float lmax = -1e30f;
        for (int j = 0; j < 16; j++) {
            const int jj = c0 + j;
            vals[j] = Ss[row][jj] * scale + bmrow[jj];
            lmax = fmaxf(lmax, vals[j]);
        }
        lmax = fmaxf(lmax, __shfl_xor(lmax, 1, 32));
        lmax = fmaxf(lmax, __shfl_xor(lmax, 2, 32));
        float lsum = 0.0f;
        for (int j = 0; j < 16; j++) {
            vals[j] = __expf(vals[j] - lmax);
            lsum += vals[j];
        }
        lsum += __shfl_xor(lsum, 1, 32);
        lsum += __shfl_xor(lsum, 2, 32);
        const float rn = 1.0f / lsum;
        for (int j = 0; j < 16; j++)
            Ps[row][c0 + j] = f2bf(vals[j] * rn);
    }
    __syncthreads();

    // ---- O = P * V : 8 tiles of 16x16, 1 per wave, K=64 in two steps ----
    {
        const int tm = wv >> 1, tn = wv & 1;
        v8f c; for (int i = 0; i < 8; i++) c[i] = 0.0f;
        for (int ks = 0; ks < 2; ks++) {
            const unsigned short* pp = &Ps[tm * 16 + lm][ks * 32];
            v16bf a  = load_frag(pp + hi16 * 8, pp + hi16 * 8 + 16);
            const unsigned short* pv = &Vt[tn * 16 + lm][ks * 32];
            v16bf bb = load_frag(pv + hi16 * 8, pv + hi16 * 8 + 16);
            c = __builtin_amdgcn_wmma_f32_16x16x32_bf16(
                    false, a, false, bb, (short)0, c, false, false);
        }
        for (int i = 0; i < 8; i++) {
            const int m = tm * 16 + hi16 * 8 + i;
            if (m < NTOK)
                attnout[((size_t)b * NTOK + m) * DIM + h * HD + tn * 16 + lm] =
                    f2bf(c[i]);
        }
    }
}

// ---------------------------------------------------------------------------
extern "C" void kernel_launch(void* const* d_in, const int* in_sizes, int n_in,
                              void* d_out, int out_size, void* d_ws, size_t ws_size,
                              hipStream_t stream) {
    (void)in_sizes; (void)n_in; (void)out_size; (void)ws_size;
    const float* x         = (const float*)d_in[0];
    const float* mask      = (const float*)d_in[1];
    const float* qkv_w     = (const float*)d_in[2];   // [1152][384]
    const float* qkv_b     = (const float*)d_in[3];
    const float* proj_w    = (const float*)d_in[4];   // [384][384]
    const float* proj_b    = (const float*)d_in[5];
    const float* rpb       = (const float*)d_in[6];
    const int*   rel_index = (const int*)d_in[7];

    // workspace carve-up (bf16 buffers first, fp32 table last)
    unsigned short* qkvb    = (unsigned short*)d_ws;                       // 255 MB
    unsigned short* attnout = qkvb    + (size_t)MROWS * (3 * DIM);         //  85 MB
    unsigned short* xb      = attnout + (size_t)MROWS * DIM;               //  85 MB
    unsigned short* wqkvb   = xb      + (size_t)MROWS * DIM;               // 0.9 MB
    unsigned short* wprojb  = wqkvb   + (size_t)(3 * DIM) * DIM;           // 0.3 MB
    float*          bm      = (float*)(wprojb + (size_t)DIM * DIM);        //  13 MB

    // 0) one-shot conversions + fused bias/mask table
    {
        int n8;
        n8 = (MROWS * DIM) / 8;
        cvt_f32_bf16_kernel<<<(n8 + 255) / 256, 256, 0, stream>>>(x, xb, n8);
        n8 = (3 * DIM * DIM) / 8;
        cvt_f32_bf16_kernel<<<(n8 + 255) / 256, 256, 0, stream>>>(qkv_w, wqkvb, n8);
        n8 = (DIM * DIM) / 8;
        cvt_f32_bf16_kernel<<<(n8 + 255) / 256, 256, 0, stream>>>(proj_w, wprojb, n8);
        build_bias_mask_kernel<<<NWMASK * HEADS, 256, 0, stream>>>(
            mask, rpb, rel_index, bm);
    }

    // 1) QKV projection (bf16 x bf16 -> bf16)
    gemm_bias_kernel<true>
        <<<dim3(MROWS / 64, (3 * DIM) / 384), 256, 0, stream>>>(
            xb, wqkvb, qkv_b, qkvb, MROWS, 3 * DIM, DIM);

    // 2) windowed attention with prompt tokens
    attn_kernel<<<dim3(HEADS, BTOT), 256, 0, stream>>>(qkvb, bm, attnout);

    // 3) output projection (bf16 x bf16 -> fp32)
    gemm_bias_kernel<false>
        <<<dim3(MROWS / 64, DIM / 384), 256, 0, stream>>>(
            attnout, wprojb, proj_b, d_out, MROWS, DIM, DIM);
}
